// LinkProp_Encoder_35003983462547
// MI455X (gfx1250) — compile-verified, matrix operand order checked by reading, not run on hardware
//
#include <hip/hip_runtime.h>
#include <hip/hip_bf16.h>

// LinkProp encoder, fused per-round kernel for gfx1250 (MI455X).
// u_next = S @ i_k ; i_next = S^T @ u_k from ONE pass over S per round.
// S chunk staged in LDS pre-swizzled into WMMA A-fragment layout (both
// orientations); embeddings kept globally in B-fragment order -> all WMMA
// operands load as wide b64/b128 accesses, no scalar gathers.

#define U_NUM 8192
#define I_NUM 16384
#define EMB   32
#define R_ROUNDS 3

#define BI  256            // items (S columns) owned per block
#define KC  32             // u rows (S rows) streamed per chunk
#define SEG 4              // split of the U stream across blockIdx.y

typedef __attribute__((ext_vector_type(4)))  _Float16 v4h;
typedef __attribute__((ext_vector_type(8)))  _Float16 v8h;
typedef __attribute__((ext_vector_type(16))) _Float16 v16h;
typedef __attribute__((ext_vector_type(8)))  float    v8f;

// ISA 7.12.2, 16-bit A-matrix 16x32: half h of lane (m, khalf) holds
// K = h + 8*khalf (h<8) or K = 16 + (h-8) + 8*khalf (h>=8).
// Inverse: given K=kk: khalf=(kk>>3)&1, h=(kk&7)|(((kk>>4)&1)<<3).

// B-fragment-order index for embedding row r, column e (lane-major, 32B/lane):
__device__ __forceinline__ size_t bperm_index(int r, int e) {
    const int g     = r >> 5;                 // group of 32 K-rows
    const int kk    = r & 31;
    const int khalf = (kk >> 3) & 1;
    const int h     = (kk & 7) | (((kk >> 4) & 1) << 3);
    const int nt    = e >> 4;
    const int lane  = (e & 15) | (khalf << 4);
    return (size_t)g * 1024 + nt * 512 + lane * 16 + h;
}

// Load one A fragment from quad-major LDS layout: conflict-free b64 loads
// (compiler fuses them into ds_load_2addr_b64 pairs).
__device__ __forceinline__ v16h load_frag_lds(const _Float16* base) {
    const v4h q0 = *reinterpret_cast<const v4h*>(base);
    const v4h q1 = *reinterpret_cast<const v4h*>(base + 128);
    const v4h q2 = *reinterpret_cast<const v4h*>(base + 256);
    const v4h q3 = *reinterpret_cast<const v4h*>(base + 384);
    const v8h lo = __builtin_shufflevector(q0, q1, 0,1,2,3,4,5,6,7);
    const v8h hi = __builtin_shufflevector(q2, q3, 0,1,2,3,4,5,6,7);
    return __builtin_shufflevector(lo, hi, 0,1,2,3,4,5,6,7,8,9,10,11,12,13,14,15);
}

__global__ __launch_bounds__(256)
void linkprop_round_kernel(const float* __restrict__ S,
                           const _Float16* __restrict__ uk16,   // B-fragment order
                           const _Float16* __restrict__ ik16,   // B-fragment order
                           float* __restrict__ u_next,
                           float* __restrict__ i_next)
{
    // A-fragment-order S tiles: [slice][mtile][h>>2][lane][h&3]
    __shared__ __attribute__((aligned(16))) _Float16 sA [8 * 1024];  // S   (u x i) 16 KB
    __shared__ __attribute__((aligned(16))) _Float16 sAT[8 * 1024];  // S^T (i x u) 16 KB
    __shared__ __attribute__((aligned(16))) float    u_red[KC * EMB];

    const int tid   = threadIdx.x;
    const int lane  = tid & 31;
    const int wave  = tid >> 5;
    const int i0    = blockIdx.x * BI;
    const int useg  = blockIdx.y;
    const int m     = lane & 15;
    const int khalf = lane >> 4;

    for (int idx = tid; idx < KC * EMB; idx += 256) u_red[idx] = 0.0f;

    // Per-wave i_next accumulators: wave's 32 item rows x 32 cols.
    v8f i_acc[2][2];
    #pragma unroll
    for (int mt = 0; mt < 2; ++mt)
        #pragma unroll
        for (int nt = 0; nt < 2; ++nt) { v8f z = {}; i_acc[mt][nt] = z; }

    // u-side work partition: wave&3 -> output quarter (mt,nt); wave>>2 -> which
    // half of the 256-item K' range this wave chains over (4 slices of 32).
    const int q_mt  = wave & 1;
    const int q_nt  = (wave >> 1) & 1;
    const int s_beg = (wave >> 2) * 4;       // first item slice of this wave's K' half

    // Hoisted B fragments for the u-side GEMM: the 4 item-slice fragments at
    // this wave's nt quarter (invariant over the whole U stream).
    v16h bI[4];
    #pragma unroll
    for (int j = 0; j < 4; ++j) {
        const _Float16* p = ik16 + (size_t)((i0 >> 5) + s_beg + j) * 1024 +
                            q_nt * 512 + lane * 16;
        bI[j] = *reinterpret_cast<const v16h*>(p);
    }

    const int ub_beg = useg * (U_NUM / SEG);
    const int ub_end = ub_beg + (U_NUM / SEG);

    // Register prefetch of two 4x4 micro-tiles of S per thread.
    // micro-tile id -> ur = (id>>6)*4 (u row), ic = (id&63)*4 (item col).
    float pf[2][4][4];
    #pragma unroll
    for (int p = 0; p < 2; ++p) {
        const int id = tid + p * 256;
        const int ur = (id >> 6) << 2;
        const int ic = (id & 63) << 2;
        #pragma unroll
        for (int du = 0; du < 4; ++du) {
            const float4 v = *reinterpret_cast<const float4*>(
                S + (size_t)(ub_beg + ur + du) * I_NUM + i0 + ic);
            pf[p][du][0] = v.x; pf[p][du][1] = v.y; pf[p][du][2] = v.z; pf[p][du][3] = v.w;
        }
    }

    for (int ub = ub_beg; ub < ub_end; ub += KC) {
        // ---- Scatter prefetched micro-tiles into both fragment layouts (b64 stores).
        #pragma unroll
        for (int p = 0; p < 2; ++p) {
            const int id = tid + p * 256;
            const int ur = (id >> 6) << 2;
            const int ic = (id & 63) << 2;
            const int wv = ic >> 5;
            // u-side (A = S[u,i]): per u-row, one quad of 4 item-K values.
            {
                const int khalf_i = (ic >> 3) & 1;
                const int h0      = (ic & 7) | (((ic >> 4) & 1) << 3);
                #pragma unroll
                for (int du = 0; du < 4; ++du) {
                    const int u  = ur + du;
                    const int off = wv * 1024 + (u >> 4) * 512 + (h0 >> 2) * 128 +
                                    (((u & 15) | (khalf_i << 4)) << 2);
                    v4h q;
                    q[0] = (_Float16)pf[p][du][0]; q[1] = (_Float16)pf[p][du][1];
                    q[2] = (_Float16)pf[p][du][2]; q[3] = (_Float16)pf[p][du][3];
                    *reinterpret_cast<v4h*>(&sA[off]) = q;
                }
            }
            // i-side (A = S^T[i,u]): per item-col, one quad of 4 user-K values.
            {
                const int khalf_u = (ur >> 3) & 1;
                const int h0      = (ur & 7) | (((ur >> 4) & 1) << 3);
                #pragma unroll
                for (int di = 0; di < 4; ++di) {
                    const int i  = ic + di;
                    const int off = wv * 1024 + ((i >> 4) & 1) * 512 + (h0 >> 2) * 128 +
                                    (((i & 15) | (khalf_u << 4)) << 2);
                    v4h q;
                    q[0] = (_Float16)pf[p][0][di]; q[1] = (_Float16)pf[p][1][di];
                    q[2] = (_Float16)pf[p][2][di]; q[3] = (_Float16)pf[p][3][di];
                    *reinterpret_cast<v4h*>(&sAT[off]) = q;
                }
            }
        }
        __syncthreads();

        // ---- Prefetch next chunk from HBM (overlaps with WMMA + DS below).
        if (ub + KC < ub_end) {
            #pragma unroll
            for (int p = 0; p < 2; ++p) {
                const int id = tid + p * 256;
                const int ur = (id >> 6) << 2;
                const int ic = (id & 63) << 2;
                #pragma unroll
                for (int du = 0; du < 4; ++du) {
                    const float4 v = *reinterpret_cast<const float4*>(
                        S + (size_t)(ub + KC + ur + du) * I_NUM + i0 + ic);
                    pf[p][du][0] = v.x; pf[p][du][1] = v.y;
                    pf[p][du][2] = v.z; pf[p][du][3] = v.w;
                }
            }
        }

        // ---- B fragments for the i-side: this chunk's 32 user embeddings.
        v16h bU[2];
        {
            const _Float16* p = uk16 + (size_t)(ub >> 5) * 1024 + lane * 16;
            bU[0] = *reinterpret_cast<const v16h*>(p);
            bU[1] = *reinterpret_cast<const v16h*>(p + 512);
        }

        // ================= i-side: i_acc += S^T(tile) @ u_k(chunk) =================
        #pragma unroll
        for (int mt = 0; mt < 2; ++mt) {
            const v16h a = load_frag_lds(&sAT[wave * 1024 + mt * 512 + lane * 4]);
            i_acc[mt][0] = __builtin_amdgcn_wmma_f32_16x16x32_f16(
                false, a, false, bU[0], (short)0, i_acc[mt][0], false, false);
            i_acc[mt][1] = __builtin_amdgcn_wmma_f32_16x16x32_f16(
                false, a, false, bU[1], (short)0, i_acc[mt][1], false, false);
        }

        // ====== u-side: register-chained over half the item range, one output
        // ====== quarter per wave; only a 2-way LDS reduction remains.
        {
            v8f c = {};
            #pragma unroll
            for (int j = 0; j < 4; ++j) {
                const v16h a = load_frag_lds(&sA[(s_beg + j) * 1024 + q_mt * 512 + lane * 4]);
                c = __builtin_amdgcn_wmma_f32_16x16x32_f16(
                    false, a, false, bI[j], (short)0, c, false, false);
            }
            #pragma unroll
            for (int v = 0; v < 8; ++v) {
                const int u_l = q_mt * 16 + v + (khalf << 3);
                const int e   = q_nt * 16 + m;
                atomicAdd(&u_red[u_l * EMB + e], c[v]);
            }
        }
        __syncthreads();

        // Flush 32x32 u partial to the L2-resident accumulator and re-zero.
        for (int idx = tid; idx < KC * EMB; idx += 256) {
            atomicAdd(&u_next[(size_t)ub * EMB + idx], u_red[idx]);
            u_red[idx] = 0.0f;
        }
        __syncthreads();
    }

    // Commit i_next partials (atomic only across the SEG u-segments).
    #pragma unroll
    for (int mt = 0; mt < 2; ++mt)
        #pragma unroll
        for (int nt = 0; nt < 2; ++nt)
            #pragma unroll
            for (int v = 0; v < 8; ++v) {
                const int i_row = i0 + wave * 32 + mt * 16 + v + (khalf << 3);
                const int e     = nt * 16 + m;
                atomicAdd(&i_next[(size_t)i_row * EMB + e], i_acc[mt][nt][v]);
            }
}

__global__ void linkprop_init_kernel(const float* __restrict__ ue,
                                     const float* __restrict__ ie,
                                     float* __restrict__ u_sum, float* __restrict__ i_sum,
                                     _Float16* __restrict__ uk16, _Float16* __restrict__ ik16,
                                     float* __restrict__ u_next, float* __restrict__ i_next)
{
    const int idx = blockIdx.x * blockDim.x + threadIdx.x;
    if (idx < U_NUM * EMB) {
        const float v = ue[idx];
        u_sum[idx] = v;
        uk16[bperm_index(idx >> 5, idx & 31)] = (_Float16)v;
        u_next[idx] = 0.0f;
    }
    if (idx < I_NUM * EMB) {
        const float v = ie[idx];
        i_sum[idx] = v;
        ik16[bperm_index(idx >> 5, idx & 31)] = (_Float16)v;
        i_next[idx] = 0.0f;
    }
}

__global__ void linkprop_update_kernel(float* __restrict__ u_sum, float* __restrict__ i_sum,
                                       _Float16* __restrict__ uk16, _Float16* __restrict__ ik16,
                                       float* __restrict__ u_next, float* __restrict__ i_next)
{
    const int idx = blockIdx.x * blockDim.x + threadIdx.x;
    if (idx < U_NUM * EMB) {
        const float v = u_next[idx];
        u_sum[idx] += v;
        uk16[bperm_index(idx >> 5, idx & 31)] = (_Float16)v;
        u_next[idx] = 0.0f;
    }
    if (idx < I_NUM * EMB) {
        const float v = i_next[idx];
        i_sum[idx] += v;
        ik16[bperm_index(idx >> 5, idx & 31)] = (_Float16)v;
        i_next[idx] = 0.0f;
    }
}

__global__ void linkprop_scale_kernel(float* __restrict__ out, int n)
{
    const int idx = blockIdx.x * blockDim.x + threadIdx.x;
    if (idx < n) out[idx] *= 0.25f;   // 1/(R+1)
}

extern "C" void kernel_launch(void* const* d_in, const int* in_sizes, int n_in,
                              void* d_out, int out_size, void* d_ws, size_t ws_size,
                              hipStream_t stream) {
    const float* user_emb = (const float*)d_in[0];   // [8192, 32]
    const float* item_emb = (const float*)d_in[1];   // [16384, 32]
    const float* S        = (const float*)d_in[2];   // [8192, 16384]

    float* u_sum = (float*)d_out;                    // outputs: u_sum, then i_sum
    float* i_sum = u_sum + (size_t)U_NUM * EMB;

    // Workspace: uk16(512K) | ik16(1M) | u_next(1M) | i_next(2M) = 4.5 MB
    char* ws = (char*)d_ws;
    _Float16* uk16 = (_Float16*)ws;
    _Float16* ik16 = (_Float16*)(ws + (size_t)U_NUM * EMB * sizeof(_Float16));
    float* u_next  = (float*)(ws + (size_t)(U_NUM + I_NUM) * EMB * sizeof(_Float16));
    float* i_next  = u_next + (size_t)U_NUM * EMB;

    const int nMax = I_NUM * EMB;
    linkprop_init_kernel<<<(nMax + 255) / 256, 256, 0, stream>>>(
        user_emb, item_emb, u_sum, i_sum, uk16, ik16, u_next, i_next);

    dim3 grid(I_NUM / BI, SEG);                      // 64 item panels x 4 u-segments
    for (int r = 0; r < R_ROUNDS; ++r) {
        linkprop_round_kernel<<<grid, 256, 0, stream>>>(S, uk16, ik16, u_next, i_next);
        linkprop_update_kernel<<<(nMax + 255) / 256, 256, 0, stream>>>(
            u_sum, i_sum, uk16, ik16, u_next, i_next);
    }

    const int nOut = (U_NUM + I_NUM) * EMB;
    linkprop_scale_kernel<<<(nOut + 255) / 256, 256, 0, stream>>>(u_sum, nOut);
}
